// FreeConvNetwork_69827578298832
// MI455X (gfx1250) — matrix operand order repaired and checked by reading, not compiled, not run on hardware
//
#include <hip/hip_runtime.h>
#include <hip/hip_bf16.h>

// ---------------------------------------------------------------------------
// FreeConvNetwork on MI455X (gfx1250).
// All GEMM stages use v_wmma_f32_16x16x32_f16 (f16 mul, f32 accumulate).
// Locally-connected layers = per-spatial-location GEMMs (M=256 batch,
// K=C*6*6 patch, N=out-channels). Patch gathers are branchless, use
// compile-time __constant__ offset tables, and index uniform base pointers
// with 32-bit offsets so loads lower to saddr+voffset (no 64-bit VALU math).
// Weights are re-packed to f16 [loc][col][k] so B fragments are 2x16B loads.
// ---------------------------------------------------------------------------

typedef __attribute__((ext_vector_type(16))) _Float16 v16h;
typedef __attribute__((ext_vector_type(8)))  _Float16 v8h;
typedef __attribute__((ext_vector_type(8)))  float    v8f;

// ---- compile-time patch-offset tables (p = c*36 + kh*6 + kw) --------------
struct OffsTab1 { int v[128]; };
static constexpr OffsTab1 make_offs1() {
    OffsTab1 t{};
    for (int p = 0; p < 128; ++p) {
        const int q  = p < 108 ? p : 107;          // clamp pad -> safe addr
        const int c  = q / 36, rem = q % 36;
        const int kh = rem / 6, kw = rem % 6;
        t.v[p] = (c * 218 + kh) * 178 + kw;        // offset within x image
    }
    return t;
}
__constant__ OffsTab1 OFFS1 = make_offs1();

struct OffsTab2 { int v[288]; };
static constexpr OffsTab2 make_offs2() {
    OffsTab2 t{};
    for (int p = 0; p < 288; ++p) {
        const int c  = p / 36, rem = p % 36;
        const int kh = rem / 6, kw = rem % 6;
        t.v[p] = (c * 54 + kh) * 44 + kw;          // offset within h1 image
    }
    return t;
}
__constant__ OffsTab2 OFFS2 = make_offs2();

// ---- ISA 7.12.2 fragment helpers (wave32) ---------------------------------
// A (16x32 f16): row = lane&15; j=0..7 -> k = khalf*8+j ; j=8..15 -> k = 16+khalf*8+(j-8)
// B (32x16 f16): col = lane&15; same k order as A (j=0..7 -> k = khalf*8+j ; j=8..15 -> k = 16+khalf*8+(j-8))
__device__ __forceinline__ v16h cat8(v8h lo, v8h hi) {
    v16h r;
#pragma unroll
    for (int j = 0; j < 8; ++j) { r[j] = lo[j]; r[j + 8] = hi[j]; }
    return r;
}
__device__ __forceinline__ v8h ld8h(const _Float16* p, int idx) {
    return *(const v8h*)(p + idx);
}

// ======================= weight repack kernels (f16) =======================
// wB1: [loc 2376][col 16][k 128], zero pad col>=8, k>=108
__global__ __launch_bounds__(256) void prep_w1(const float* __restrict__ w1,
                                               _Float16* __restrict__ wB1) {
    const int i = blockIdx.x * 256 + threadIdx.x;
    if (i >= 2376 * 16 * 128) return;
    const int pk = i & 127, rest = i >> 7;
    const int col = rest & 15, loc = rest >> 4;
    const int ho = loc / 44, wo = loc % 44;
    float v = 0.f;
    if (col < 8 && pk < 108) v = w1[((col * 108 + pk) * 54 + ho) * 44 + wo];
    wB1[i] = (_Float16)v;
}
// wB2: [loc 130][col 16][k 288] (no padding needed)
__global__ __launch_bounds__(256) void prep_w2(const float* __restrict__ w2,
                                               _Float16* __restrict__ wB2) {
    const int i = blockIdx.x * 256 + threadIdx.x;
    if (i >= 130 * 16 * 288) return;
    const int pk = i % 288, rest = i / 288;
    const int col = rest & 15, loc = rest >> 4;
    const int ho = loc / 10, wo = loc % 10;
    wB2[i] = (_Float16)w2[((col * 288 + pk) * 13 + ho) * 10 + wo];
}
// wF1: [n 304][k 480], zero pad n>=300
__global__ __launch_bounds__(256) void prep_fc1(const float* __restrict__ w,
                                                _Float16* __restrict__ wF1) {
    const int i = blockIdx.x * 256 + threadIdx.x;
    if (i >= 304 * 480) return;
    const int k = i % 480, n = i / 480;
    wF1[i] = (_Float16)(n < 300 ? w[n * 480 + k] : 0.f);
}
// wF2: [n 208][k 320], zero pad n>=200, k>=300
__global__ __launch_bounds__(256) void prep_fc2(const float* __restrict__ w,
                                                _Float16* __restrict__ wF2) {
    const int i = blockIdx.x * 256 + threadIdx.x;
    if (i >= 208 * 320) return;
    const int k = i % 320, n = i / 320;
    wF2[i] = (_Float16)((n < 200 && k < 300) ? w[n * 300 + k] : 0.f);
}

// ======================= LC1: (256,3,218,178) -> (256,8,54,44) =============
// 4 waves/block = 4 m-tiles of the same location (shared B lines -> WGP$ hits)
__global__ __launch_bounds__(128) void lc1_wmma(
    const float* __restrict__ x, const _Float16* __restrict__ wB1,
    const float* __restrict__ b1, _Float16* __restrict__ h1)
{
    const int loc   = blockIdx.x;                       // 0..2375
    const int mtile = blockIdx.y * 4 + (threadIdx.x >> 5);  // 0..15
    const int lane  = threadIdx.x & 31;
    const int ho = loc / 44, wo = loc % 44;
    const int arow  = lane & 15;
    const int col   = lane & 15;
    const int khalf = lane >> 4;
    const int bglob = mtile * 16 + arow;

    const int xbase = bglob * 116412 + (ho * 4) * 178 + (wo * 4);  // 32-bit idx
    const int wbase = (loc * 16 + col) * 128 + (khalf << 3);

    v8f acc = {};
#pragma unroll
    for (int ks = 0; ks < 4; ++ks) {
        const int kbase = ks * 32;
        // gather offsets: 4x int4 from constant table (contiguous runs)
        union { int4 q[4]; int s[16]; } u;
        const int4* t0 = (const int4*)(OFFS1.v + kbase + (khalf << 3));
        const int4* t1 = (const int4*)(OFFS1.v + kbase + 16 + (khalf << 3));
        u.q[0] = t0[0]; u.q[1] = t0[1]; u.q[2] = t1[0]; u.q[3] = t1[1];
        // branchless A gather: uniform base + 32-bit voffset per element
        float va[16];
#pragma unroll
        for (int j = 0; j < 16; ++j) va[j] = x[xbase + u.s[j]];
        v16h a;
#pragma unroll
        for (int j = 0; j < 16; ++j) {
            const int pk = kbase + ((j >> 3) << 4) + (khalf << 3) + (j & 7);
            a[j] = (_Float16)(pk < 108 ? va[j] : 0.f);
        }
        // B: two 16B vector loads of pre-packed f16 weights
        const v16h b = cat8(ld8h(wB1, wbase + kbase), ld8h(wB1, wbase + kbase + 16));
        acc = __builtin_amdgcn_wmma_f32_16x16x32_f16(false, a, false, b,
                                                     (short)0, acc, false, false);
    }
    if (col < 8) {
        const float bias = b1[(col * 54 + ho) * 44 + wo];
#pragma unroll
        for (int r = 0; r < 8; ++r) {
            const int bo = mtile * 16 + r + (khalf << 3);
            float v = acc[r] + bias;
            v = v > 0.f ? v : 0.f;
            h1[((bo * 8 + col) * 54 + ho) * 44 + wo] = (_Float16)v;
        }
    }
}

// ======================= LC2: (256,8,54,44) -> (256,16,13,10) ==============
__global__ __launch_bounds__(128) void lc2_wmma(
    const _Float16* __restrict__ h1, const _Float16* __restrict__ wB2,
    const float* __restrict__ b2, float* __restrict__ h2)
{
    const int loc   = blockIdx.x;                       // 0..129
    const int mtile = blockIdx.y * 4 + (threadIdx.x >> 5);
    const int lane  = threadIdx.x & 31;
    const int ho = loc / 10, wo = loc % 10;
    const int arow  = lane & 15;
    const int col   = lane & 15;
    const int khalf = lane >> 4;
    const int bglob = mtile * 16 + arow;

    const int hbase = bglob * 19008 + (ho * 4) * 44 + (wo * 4);
    const int wbase = (loc * 16 + col) * 288 + (khalf << 3);

    v8f acc = {};
    for (int ks = 0; ks < 9; ++ks) {         // K = 288 exact, no masks anywhere
        const int kbase = ks * 32;
        union { int4 q[4]; int s[16]; } u;
        const int4* t0 = (const int4*)(OFFS2.v + kbase + (khalf << 3));
        const int4* t1 = (const int4*)(OFFS2.v + kbase + 16 + (khalf << 3));
        u.q[0] = t0[0]; u.q[1] = t0[1]; u.q[2] = t1[0]; u.q[3] = t1[1];
        v16h a;
#pragma unroll
        for (int j = 0; j < 16; ++j) a[j] = h1[hbase + u.s[j]];
        const v16h b = cat8(ld8h(wB2, wbase + kbase), ld8h(wB2, wbase + kbase + 16));
        acc = __builtin_amdgcn_wmma_f32_16x16x32_f16(false, a, false, b,
                                                     (short)0, acc, false, false);
    }
    {
        const float bias = b2[(col * 13 + ho) * 10 + wo];
#pragma unroll
        for (int r = 0; r < 8; ++r) {
            const int bo = mtile * 16 + r + (khalf << 3);
            float v = acc[r] + bias;
            v = v > 0.f ? v : 0.f;
            h2[((bo * 16 + col) * 13 + ho) * 10 + wo] = v;
        }
    }
}

// ======================= MaxPool2d(2,floor) + flatten -> (256,480) f16 =====
__global__ __launch_bounds__(256) void pool_flatten(const float* __restrict__ h2,
                                                    _Float16* __restrict__ pooled)
{
    const int idx = blockIdx.x * 256 + threadIdx.x;   // b*480 + f
    if (idx >= 256 * 480) return;
    const int b = idx / 480, f = idx - b * 480;
    const int o = f / 30, rem = f - o * 30;
    const int ph = rem / 5, pw = rem - ph * 5;
    const int base = ((b * 16 + o) * 13 + ph * 2) * 10 + pw * 2;
    const float m = fmaxf(fmaxf(h2[base], h2[base + 1]),
                          fmaxf(h2[base + 10], h2[base + 11]));
    pooled[idx] = (_Float16)m;
}

// ======================= FC1: (256,480)x(480,300)^T + ReLU -> h3[256][320] =
__global__ __launch_bounds__(128) void fc1_wmma(
    const _Float16* __restrict__ A, const _Float16* __restrict__ wF1,
    const float* __restrict__ bias, _Float16* __restrict__ h3)
{
    const int ntile = blockIdx.x;                       // 0..19 (padded 320)
    const int mtile = blockIdx.y * 4 + (threadIdx.x >> 5);
    const int lane  = threadIdx.x & 31;
    const int arow  = lane & 15;
    const int col   = lane & 15;
    const int khalf = lane >> 4;
    const int bglob = mtile * 16 + arow;
    const int n     = ntile * 16 + col;

    const int abase = bglob * 480 + (khalf << 3);
    const int wbase = (n < 304 ? n : 303) * 480 + (khalf << 3);

    v8f acc = {};
    for (int ks = 0; ks < 15; ++ks) {        // K = 480 exact
        const int kbase = ks * 32;
        const v16h a = cat8(ld8h(A, abase + kbase), ld8h(A, abase + kbase + 16));
        const v16h b = cat8(ld8h(wF1, wbase + kbase), ld8h(wF1, wbase + kbase + 16));
        acc = __builtin_amdgcn_wmma_f32_16x16x32_f16(false, a, false, b,
                                                     (short)0, acc, false, false);
    }
    const float bs = bias[n < 300 ? n : 0];
#pragma unroll
    for (int r = 0; r < 8; ++r) {
        const int bo = mtile * 16 + r + (khalf << 3);
        float v = acc[r] + bs;
        v = v > 0.f ? v : 0.f;
        h3[bo * 320 + n] = (_Float16)(n < 300 ? v : 0.f);  // zero pad cols
    }
}

// ======================= FC2: (256,320pad)x(320,208pad)^T ------------------
__global__ __launch_bounds__(128) void fc2_wmma(
    const _Float16* __restrict__ A, const _Float16* __restrict__ wF2,
    const float* __restrict__ bias, float* __restrict__ logits)
{
    const int ntile = blockIdx.x;                       // 0..12
    const int mtile = blockIdx.y * 4 + (threadIdx.x >> 5);
    const int lane  = threadIdx.x & 31;
    const int arow  = lane & 15;
    const int col   = lane & 15;
    const int khalf = lane >> 4;
    const int bglob = mtile * 16 + arow;
    const int n     = ntile * 16 + col;

    const int abase = bglob * 320 + (khalf << 3);
    const int wbase = n * 320 + (khalf << 3);           // n<208 always valid

    v8f acc = {};
    for (int ks = 0; ks < 10; ++ks) {        // K padded to 320, zeros in pad
        const int kbase = ks * 32;
        const v16h a = cat8(ld8h(A, abase + kbase), ld8h(A, abase + kbase + 16));
        const v16h b = cat8(ld8h(wF2, wbase + kbase), ld8h(wF2, wbase + kbase + 16));
        acc = __builtin_amdgcn_wmma_f32_16x16x32_f16(false, a, false, b,
                                                     (short)0, acc, false, false);
    }
    if (n < 200) {
        const float bs = bias[n];
#pragma unroll
        for (int r = 0; r < 8; ++r) {
            const int bo = mtile * 16 + r + (khalf << 3);
            logits[bo * 200 + n] = acc[r] + bs;
        }
    }
}

// ======================= Softmax over 200 classes, one wave per row ========
__global__ __launch_bounds__(32) void softmax200(
    const float* __restrict__ logits, float* __restrict__ out)
{
    const int b = blockIdx.x;
    const int lane = threadIdx.x;
    float m = -3.4e38f;
    for (int i = lane; i < 200; i += 32) m = fmaxf(m, logits[b * 200 + i]);
#pragma unroll
    for (int off = 16; off > 0; off >>= 1) m = fmaxf(m, __shfl_xor(m, off, 32));
    float s = 0.f;
    for (int i = lane; i < 200; i += 32) s += __expf(logits[b * 200 + i] - m);
#pragma unroll
    for (int off = 16; off > 0; off >>= 1) s += __shfl_xor(s, off, 32);
    const float inv = 1.f / s;
    for (int i = lane; i < 200; i += 32)
        out[b * 200 + i] = __expf(logits[b * 200 + i] - m) * inv;
}

// ---------------------------------------------------------------------------
extern "C" void kernel_launch(void* const* d_in, const int* in_sizes, int n_in,
                              void* d_out, int out_size, void* d_ws, size_t ws_size,
                              hipStream_t stream)
{
    (void)in_sizes; (void)n_in; (void)out_size; (void)ws_size;
    const float* x     = (const float*)d_in[0];
    const float* w1    = (const float*)d_in[1];
    const float* b1    = (const float*)d_in[2];
    const float* w2    = (const float*)d_in[3];
    const float* b2    = (const float*)d_in[4];
    const float* fc1_w = (const float*)d_in[5];
    const float* fc1_b = (const float*)d_in[6];
    const float* fc2_w = (const float*)d_in[7];
    const float* fc2_b = (const float*)d_in[8];
    float* out = (float*)d_out;

    // Workspace carve-up (~23.8 MB, every region fully overwritten per call).
    char* ws = (char*)d_ws;
    _Float16* h1     = (_Float16*)(ws);              //  9,732,096 B  (256*8*54*44 f16)
    float*    h2     = (float*)   (ws + 9732096);    //  2,129,920 B  (256*16*13*10 f32)
    _Float16* pooled = (_Float16*)(ws + 11862016);   //    245,760 B  (256*480 f16)
    _Float16* h3     = (_Float16*)(ws + 12107776);   //    163,840 B  (256*320 f16, padded)
    float*    logit  = (float*)   (ws + 12271616);   //    204,800 B  (256*200 f32)
    _Float16* wB1    = (_Float16*)(ws + 12476416);   //  9,732,096 B  (2376*16*128 f16)
    _Float16* wB2    = (_Float16*)(ws + 22208512);   //  1,198,080 B  (130*16*288 f16)
    _Float16* wF1    = (_Float16*)(ws + 23406592);   //    291,840 B  (304*480 f16)
    _Float16* wF2    = (_Float16*)(ws + 23698432);   //    133,120 B  (208*320 f16)
    // total: 23,831,552 B

    prep_w1 <<<dim3((2376 * 16 * 128 + 255) / 256), dim3(256), 0, stream>>>(w1, wB1);
    prep_w2 <<<dim3((130 * 16 * 288 + 255) / 256),  dim3(256), 0, stream>>>(w2, wB2);
    prep_fc1<<<dim3((304 * 480 + 255) / 256),       dim3(256), 0, stream>>>(fc1_w, wF1);
    prep_fc2<<<dim3((208 * 320 + 255) / 256),       dim3(256), 0, stream>>>(fc2_w, wF2);

    lc1_wmma<<<dim3(54 * 44, 4), dim3(128), 0, stream>>>(x, wB1, b1, h1);
    lc2_wmma<<<dim3(13 * 10, 4), dim3(128), 0, stream>>>(h1, wB2, b2, h2);
    pool_flatten<<<dim3((256 * 480 + 255) / 256), dim3(256), 0, stream>>>(h2, pooled);
    fc1_wmma<<<dim3(20, 4), dim3(128), 0, stream>>>(pooled, wF1, fc1_b, h3);
    fc2_wmma<<<dim3(13, 4), dim3(128), 0, stream>>>(h3, wF2, fc2_b, logit);
    softmax200<<<dim3(256), dim3(32), 0, stream>>>(logit, out);
}
